// DirectionalConv_60748017434831
// MI455X (gfx1250) — matrix operand order repaired and checked by reading, not compiled, not run on hardware
//
#include <hip/hip_runtime.h>

// ---------------------------------------------------------------------------
// DirectionalConv for MI455X (gfx1250, wave32, WMMA, async LDS staging)
// ---------------------------------------------------------------------------

typedef __attribute__((ext_vector_type(16))) __bf16 v16bf;
typedef __attribute__((ext_vector_type(8)))  float  v8f;

#define B_    8
#define C_    64
#define H_    256
#define W_    256
#define OC_   64
#define KB_   8

#define TILE_H  8
#define TILE_W  16
#define PATCH_H 10
#define PATCH_W 18
#define NPATCH  (PATCH_H * PATCH_W)

__device__ __forceinline__ unsigned short f2bf(float f) {
  unsigned int u = __float_as_uint(f);
  return (unsigned short)((u + 0x7FFFu + ((u >> 16) & 1u)) >> 16);  // RNE
}

// mask one 32-bit word (2 bf16 channels) by per-channel bin equality
__device__ __forceinline__ unsigned int mask2(unsigned int xw, unsigned int qw,
                                              int byteoff, unsigned int kk) {
  unsigned int q0 = (qw >> (byteoff * 8)) & 0xFFu;
  unsigned int q1 = (qw >> (byteoff * 8 + 8)) & 0xFFu;
  unsigned int m  = (q0 == kk ? 0x0000FFFFu : 0u) | (q1 == kk ? 0xFFFF0000u : 0u);
  return xw & m;
}

// ---------------------------------------------------------------------------
// Stage 1: Sobel -> atan2 -> bin q; x -> bf16, both in [b][h][w][c] layout
// ---------------------------------------------------------------------------
__global__ __launch_bounds__(256)
void prep_kernel(const float* __restrict__ x,
                 unsigned short* __restrict__ xbf,
                 unsigned char* __restrict__ qv) {
  const int idx = blockIdx.x * blockDim.x + threadIdx.x;  // over B*H*W
  const int w = idx % W_;
  const int h = (idx / W_) % H_;
  const int b = idx / (W_ * H_);
  const float PI = 3.14159265358979323846f;
  const float SC = (float)KB_ / (2.0f * PI);
  const long plane = (long)H_ * W_;
  const float* xb = x + (long)b * C_ * plane;
  const long o = (long)idx * C_;
  for (int c = 0; c < C_; ++c) {
    const float* xp = xb + (long)c * plane;
    float v[3][3];
#pragma unroll
    for (int i = 0; i < 3; ++i)
#pragma unroll
      for (int j = 0; j < 3; ++j) {
        int hh = h + i - 1, ww = w + j - 1;
        v[i][j] = (hh >= 0 && hh < H_ && ww >= 0 && ww < W_) ? xp[hh * W_ + ww] : 0.0f;
      }
    float gx = v[0][0] - v[0][2] + 2.f * (v[1][0] - v[1][2]) + v[2][0] - v[2][2];
    float gy = v[0][0] + 2.f * v[0][1] + v[0][2] - v[2][0] - 2.f * v[2][1] - v[2][2];
    float th = atan2f(gy, gx);
    int q = (int)((th + PI) * SC);  // >=0, trunc == floor
    q &= (KB_ - 1);
    xbf[o + c] = f2bf(v[1][1]);
    qv[o + c]  = (unsigned char)q;
  }
}

// ---------------------------------------------------------------------------
// Stage 2: repack weights into WMMA B-fragment order (bf16).
// Fragment f = ((kk*9 + tap)*2 + cc)*4 + nt ; within: lane l, half v
//   B 32x16 bf16 layout: lane<16 -> N=l, K in {0..7,16..23} ; lane>=16 -> N=l-16, K +8
// ---------------------------------------------------------------------------
__global__ __launch_bounds__(256)
void wprep_kernel(const float* __restrict__ W,
                  unsigned short* __restrict__ Wb) {
  const int idx = blockIdx.x * blockDim.x + threadIdx.x;
  const int total = KB_ * 9 * 2 * 4 * 512;  // 294912
  if (idx >= total) return;
  const int v  = idx & 15;
  const int l  = (idx >> 4) & 31;
  const int f  = idx >> 9;
  const int nt = f & 3;
  const int cc = (f >> 2) & 1;
  const int t  = (f >> 3) % 9;
  const int kk = f / 72;
  const int c  = cc * 32 + (v & 7) + ((l < 16) ? 0 : 8) + ((v & 8) ? 16 : 0);
  const int o  = nt * 16 + (l & 15);
  const float wv = W[(((long)(kk * OC_ + o) * C_ + c) * 9) + t];
  Wb[idx] = f2bf(wv);
}

// ---------------------------------------------------------------------------
// Stage 3: implicit-GEMM directional conv. One WG = 8x16 pixel tile, all 64
// outputs. Wave w = output row w of the tile, M=16 pixels, N=64 (4 subtiles).
// Patch staged with CDNA5 async global->LDS DMA (ASYNCcnt); B fragments
// double-buffered over the unrolled bin loop so weight loads overlap WMMAs.
// ---------------------------------------------------------------------------
__global__ __launch_bounds__(256)
void dirconv_kernel(const unsigned short* __restrict__ xbf,
                    const unsigned char* __restrict__ qv,
                    const unsigned short* __restrict__ Wb,
                    const float* __restrict__ bias,
                    float* __restrict__ out) {
  __shared__ __align__(16) unsigned short s_x[NPATCH * C_];  // 23040 B
  __shared__ __align__(16) unsigned char  s_q[NPATCH * C_];  // 11520 B
  __shared__ __align__(16) float          s_o[OC_ * TILE_H * TILE_W];  // 32 KB

  const int tid  = threadIdx.x;
  const int wgid = blockIdx.x;
  const int tw = wgid % (W_ / TILE_W);
  const int th = (wgid / (W_ / TILE_W)) % (H_ / TILE_H);
  const int b  = wgid / ((W_ / TILE_W) * (H_ / TILE_H));
  const int h0 = th * TILE_H, w0 = tw * TILE_W;

  // hint: weights are hot and should be L2-resident
  __builtin_prefetch(Wb + ((long)(tid & 63) * 4608), 0, 3);

  // ---- stage halo patch (rows h0-1..h0+8, cols w0-1..w0+16), channel-last
  for (int p = tid; p < NPATCH; p += 256) {
    const int pr = p / PATCH_W, pc = p % PATCH_W;
    const int hh = h0 + pr - 1, ww = w0 + pc - 1;
    const bool inb = (hh >= 0 && hh < H_ && ww >= 0 && ww < W_);
    const long g = inb ? (((long)(b * H_ + hh) * W_ + ww) * C_) : 0;
    uint4* sx = (uint4*)(s_x + p * C_);
    uint4* sq = (uint4*)(s_q + p * C_);
    if (inb) {
      // async DMA: global -> LDS, tracked by ASYNCcnt (no VGPR round-trip)
      const unsigned int lx = (unsigned int)(size_t)(void*)sx;
      const unsigned int lq = (unsigned int)(size_t)(void*)sq;
      const unsigned long long gx = (unsigned long long)(size_t)(xbf + g);
      const unsigned long long gq = (unsigned long long)(size_t)(qv + g);
#pragma unroll
      for (int i = 0; i < 8; ++i)
        asm volatile("global_load_async_to_lds_b128 %0, %1, off"
                     :: "v"(lx + 16u * i), "v"(gx + 16u * i) : "memory");
#pragma unroll
      for (int i = 0; i < 4; ++i)
        asm volatile("global_load_async_to_lds_b128 %0, %1, off"
                     :: "v"(lq + 16u * i), "v"(gq + 16u * i) : "memory");
    } else {
      const uint4 z = make_uint4(0u, 0u, 0u, 0u);
#pragma unroll
      for (int i = 0; i < 8; ++i) sx[i] = z;
#pragma unroll
      for (int i = 0; i < 4; ++i) sq[i] = z;
    }
  }
  asm volatile("s_wait_asynccnt 0x0" ::: "memory");
  __syncthreads();

  const int lane    = tid & 31;
  const int wave    = tid >> 5;      // 0..7 : output row within tile
  const int laneM   = lane & 15;     // M index (pixel column)
  const int hiHalf  = lane >> 4;     // 0/1 : K-half selector for A

  v8f acc[4];
#pragma unroll
  for (int nt = 0; nt < 4; ++nt)
#pragma unroll
    for (int r = 0; r < 8; ++r) acc[nt][r] = 0.0f;

  const unsigned short* WbL = Wb + (size_t)lane * 16;  // per-lane fragment base

  for (int dy = 0; dy < 3; ++dy) {
    const int prow = wave + dy;
    for (int dx = 0; dx < 3; ++dx) {
      const int tap = dy * 3 + dx;
      const int pix = prow * PATCH_W + laneM + dx;
      for (int cc = 0; cc < 2; ++cc) {
        // A layout (16-bit A 16x32): lane<16 K in {0..7,16..23}, lane>=16 +8
        const int cA = cc * 32 + 8 * hiHalf;
        const uint4* xa = (const uint4*)(s_x + pix * C_ + cA);
        const uint4 x0 = xa[0];              // channels cA .. cA+7
        const uint4 x1 = xa[2];              // channels cA+16 .. cA+23
        const uint2* qa = (const uint2*)(s_q + pix * C_ + cA);
        const uint2 q0 = qa[0];
        const uint2 q1 = qa[2];

        // double-buffered weight fragments across the 8 orientation bins
        v16bf Bf[2][4];
#pragma unroll
        for (int nt = 0; nt < 4; ++nt)
          Bf[0][nt] = *(const v16bf*)(WbL + (size_t)((tap * 2 + cc) * 4 + nt) * 512);

#pragma unroll
        for (int kk = 0; kk < 8; ++kk) {
          const int cur = kk & 1, nxt = cur ^ 1;
          if (kk < 7) {  // issue next bin's weights before this bin's WMMAs
            const int fb = (((kk + 1) * 9 + tap) * 2 + cc) * 4;
#pragma unroll
            for (int nt = 0; nt < 4; ++nt)
              Bf[nxt][nt] = *(const v16bf*)(WbL + (size_t)(fb + nt) * 512);
          }
          union { unsigned int u[8]; v16bf v; } A;
          A.u[0] = mask2(x0.x, q0.x, 0, (unsigned)kk);
          A.u[1] = mask2(x0.y, q0.x, 2, (unsigned)kk);
          A.u[2] = mask2(x0.z, q0.y, 0, (unsigned)kk);
          A.u[3] = mask2(x0.w, q0.y, 2, (unsigned)kk);
          A.u[4] = mask2(x1.x, q1.x, 0, (unsigned)kk);
          A.u[5] = mask2(x1.y, q1.x, 2, (unsigned)kk);
          A.u[6] = mask2(x1.z, q1.y, 0, (unsigned)kk);
          A.u[7] = mask2(x1.w, q1.y, 2, (unsigned)kk);
          // 4 WMMAs share A: set reuse-A hint on the repeated ones
          acc[0] = __builtin_amdgcn_wmma_f32_16x16x32_bf16(
              false, A.v, false, Bf[cur][0], (short)0, acc[0], false, false);
          acc[1] = __builtin_amdgcn_wmma_f32_16x16x32_bf16(
              false, A.v, false, Bf[cur][1], (short)0, acc[1], true, false);
          acc[2] = __builtin_amdgcn_wmma_f32_16x16x32_bf16(
              false, A.v, false, Bf[cur][2], (short)0, acc[2], true, false);
          acc[3] = __builtin_amdgcn_wmma_f32_16x16x32_bf16(
              false, A.v, false, Bf[cur][3], (short)0, acc[3], true, false);
        }
      }
    }
  }

  // ---- epilogue: scale 1/K, add summed bias, transpose via LDS, store
#pragma unroll
  for (int nt = 0; nt < 4; ++nt) {
    const int o = nt * 16 + laneM;
    float bs = 0.0f;
#pragma unroll
    for (int k = 0; k < KB_; ++k) bs += bias[k * OC_ + o];
    bs *= 0.125f;
#pragma unroll
    for (int r = 0; r < 8; ++r) {
      const int m = r + 8 * hiHalf;  // D layout: lane>=16 holds M = r+8
      s_o[o * (TILE_H * TILE_W) + wave * TILE_W + m] = acc[nt][r] * 0.125f + bs;
    }
  }
  __syncthreads();

  // coalesced stores: 64 o-planes x 8 rows, 16 contiguous floats each
  for (int pr = tid; pr < OC_ * TILE_H; pr += 256) {
    const int o = pr / TILE_H;
    const int r = pr % TILE_H;
    const float4* src = (const float4*)(s_o + o * (TILE_H * TILE_W) + r * TILE_W);
    float4* dst = (float4*)(out + (((long)(b * OC_ + o) * H_ + (h0 + r)) * W_ + w0));
#pragma unroll
    for (int i = 0; i < 4; ++i) dst[i] = src[i];
  }
}

// ---------------------------------------------------------------------------
extern "C" void kernel_launch(void* const* d_in, const int* in_sizes, int n_in,
                              void* d_out, int out_size, void* d_ws, size_t ws_size,
                              hipStream_t stream) {
  const float* x    = (const float*)d_in[0];   // (8,64,256,256) f32
  const float* W    = (const float*)d_in[1];   // (8,64,64,3,3) f32
  const float* bias = (const float*)d_in[2];   // (8,64) f32
  float* out = (float*)d_out;                  // (8,64,256,256) f32

  char* ws = (char*)d_ws;
  const size_t nPix = (size_t)B_ * H_ * W_ * C_;          // 33,554,432
  unsigned short* xbf = (unsigned short*)ws;               // 64 MB
  unsigned char*  qv  = (unsigned char*)(ws + nPix * 2);   // 32 MB
  unsigned short* Wb  = (unsigned short*)(ws + nPix * 3);  // 576 KB

  prep_kernel<<<(B_ * H_ * W_) / 256, 256, 0, stream>>>(x, xbf, qv);
  wprep_kernel<<<(KB_ * 9 * 2 * 4 * 512 + 255) / 256, 256, 0, stream>>>(W, Wb);
  dirconv_kernel<<<B_ * (H_ / TILE_H) * (W_ / TILE_W), 256, 0, stream>>>(
      xbf, qv, Wb, bias, out);
}
